// PointNet2FPModule_11639361372333
// MI455X (gfx1250) — matrix operand order repaired and checked by reading, not compiled, not run on hardware
//
#include <hip/hip_runtime.h>

// ---------------- problem constants ----------------
#define BB   2
#define NN   16384
#define MM   4096
#define C1   128
#define C2   256
#define CIN  384      // C1 + C2
#define H1   256
#define H2   128

typedef __attribute__((ext_vector_type(16))) _Float16 v16h;
typedef __attribute__((ext_vector_type(8)))  _Float16 v8h;
typedef __attribute__((ext_vector_type(8)))  float    v8f;

typedef union { v16h v; v8h h[2]; } v16u;

// ---------------- kernel 1: weight conversion to f16 ----------------
__global__ void convert_w_kernel(const float* __restrict__ W1,
                                 const float* __restrict__ W2,
                                 _Float16* __restrict__ W1h,
                                 _Float16* __restrict__ W2h) {
    int i = blockIdx.x * blockDim.x + threadIdx.x;
    const int n1 = H1 * CIN;             // 98304
    const int n2 = H2 * H1;              // 32768
    if (i < n1) {
        W1h[i] = (_Float16)W1[i];
    } else if (i < n1 + n2) {
        int j = i - n1;
        W2h[j] = (_Float16)W2[j];
    }
}

// ---------------- kernel 2: transpose known_feats (B,C2,M)->(B,M,C2) ----------------
__global__ void transpose_kf_kernel(const float* __restrict__ kf,
                                    float* __restrict__ kft) {
    int t = blockIdx.x * blockDim.x + threadIdx.x;   // over B*C2*M, coalesced in m
    int b = t / (C2 * MM);
    int r = t - b * (C2 * MM);
    int c = r / MM;
    int m = r - c * MM;
    kft[((size_t)b * MM + m) * C2 + c] = kf[t];
}

// ---------------- kernel 3: 3-NN + inverse-distance weights ----------------
__global__ __launch_bounds__(256) void knn3_kernel(const float* __restrict__ unknown,
                                                   const float* __restrict__ known,
                                                   int*   __restrict__ idx_out,
                                                   float* __restrict__ w_out) {
    __shared__ float sx[MM];
    __shared__ float sy[MM];
    __shared__ float sz[MM];
    const int b = blockIdx.y;
    const int tid = threadIdx.x;
    const float* kb = known + (size_t)b * MM * 3;
    for (int t = tid; t < MM; t += 256) {
        sx[t] = kb[t * 3 + 0];
        sy[t] = kb[t * 3 + 1];
        sz[t] = kb[t * 3 + 2];
    }
    __syncthreads();

    const int n = blockIdx.x * 256 + tid;
    const float* u = unknown + ((size_t)b * NN + n) * 3;
    const float ux = u[0], uy = u[1], uz = u[2];

    float d0 = 3.4e38f, d1 = 3.4e38f, d2 = 3.4e38f;
    int   i0 = 0,       i1 = 0,       i2 = 0;
    for (int m = 0; m < MM; ++m) {
        float dx = ux - sx[m];
        float dy = uy - sy[m];
        float dz = uz - sz[m];
        float d  = dx * dx + dy * dy + dz * dz;
        if (d < d2) {
            if (d < d1) {
                d2 = d1; i2 = i1;
                if (d < d0) { d1 = d0; i1 = i0; d0 = d; i0 = m; }
                else        { d1 = d;  i1 = m; }
            } else { d2 = d; i2 = m; }
        }
    }
    float s0 = sqrtf(fmaxf(d0, 0.f));
    float s1 = sqrtf(fmaxf(d1, 0.f));
    float s2 = sqrtf(fmaxf(d2, 0.f));
    float r0 = 1.0f / (s0 + 1e-8f);
    float r1 = 1.0f / (s1 + 1e-8f);
    float r2 = 1.0f / (s2 + 1e-8f);
    float inv = 1.0f / (r0 + r1 + r2);
    size_t o = ((size_t)b * NN + n) * 3;
    idx_out[o + 0] = i0; idx_out[o + 1] = i1; idx_out[o + 2] = i2;
    w_out[o + 0] = r0 * inv; w_out[o + 1] = r1 * inv; w_out[o + 2] = r2 * inv;
}

// ---------------- kernel 4: interpolate + concat -> Xt (B, N, 384) f16, n-major ----------------
__global__ __launch_bounds__(128) void interp_concat_kernel(const float* __restrict__ kft,  // (B,M,C2)
                                                            const float* __restrict__ uf,   // (B,C1,N)
                                                            const int*   __restrict__ idx,
                                                            const float* __restrict__ wgt,
                                                            _Float16*    __restrict__ Xt) {
    const int b = blockIdx.y;
    const int n = blockIdx.x * 128 + threadIdx.x;
    size_t o = ((size_t)b * NN + n) * 3;
    const int   j0 = idx[o + 0], j1 = idx[o + 1], j2 = idx[o + 2];
    const float w0 = wgt[o + 0], w1 = wgt[o + 1], w2 = wgt[o + 2];
    const float* r0 = kft + ((size_t)b * MM + j0) * C2;
    const float* r1 = kft + ((size_t)b * MM + j1) * C2;
    const float* r2 = kft + ((size_t)b * MM + j2) * C2;
    _Float16* xrow = Xt + ((size_t)b * NN + n) * CIN;

    // channels 0..255: interpolated known feats
    for (int c = 0; c < C2; c += 8) {
        float4 a0 = *(const float4*)(r0 + c);
        float4 a1 = *(const float4*)(r0 + c + 4);
        float4 b0 = *(const float4*)(r1 + c);
        float4 b1 = *(const float4*)(r1 + c + 4);
        float4 c0 = *(const float4*)(r2 + c);
        float4 c1 = *(const float4*)(r2 + c + 4);
        v8h out;
        out[0] = (_Float16)(w0 * a0.x + w1 * b0.x + w2 * c0.x);
        out[1] = (_Float16)(w0 * a0.y + w1 * b0.y + w2 * c0.y);
        out[2] = (_Float16)(w0 * a0.z + w1 * b0.z + w2 * c0.z);
        out[3] = (_Float16)(w0 * a0.w + w1 * b0.w + w2 * c0.w);
        out[4] = (_Float16)(w0 * a1.x + w1 * b1.x + w2 * c1.x);
        out[5] = (_Float16)(w0 * a1.y + w1 * b1.y + w2 * c1.y);
        out[6] = (_Float16)(w0 * a1.z + w1 * b1.z + w2 * c1.z);
        out[7] = (_Float16)(w0 * a1.w + w1 * b1.w + w2 * c1.w);
        *(v8h*)(xrow + c) = out;
    }
    // channels 256..383: copy unknow_feats (coalesced across lanes per channel)
    const float* ub = uf + (size_t)b * C1 * NN + n;
    for (int c = 0; c < C1; c += 8) {
        v8h out;
        #pragma unroll
        for (int j = 0; j < 8; ++j)
            out[j] = (_Float16)ub[(size_t)(c + j) * NN];
        *(v8h*)(xrow + C2 + c) = out;
    }
}

// ---------------- kernel 5: GEMM1  H = relu(W1 * X + b1), WMMA f16 ----------------
// A = W1h (H1 x CIN) row-major; B^T = Xt (N x CIN) n-major; out Ht (B,N,H1) f16 n-major
__global__ __launch_bounds__(256) void gemm1_kernel(const _Float16* __restrict__ W1h,
                                                    const _Float16* __restrict__ Xt,
                                                    const float*    __restrict__ bias,
                                                    _Float16*       __restrict__ Ht) {
    const int b    = blockIdx.z;
    const int tid  = threadIdx.x;
    const int wave = tid >> 5, lane = tid & 31;
    const int lm   = lane & 15, lh = lane >> 4;
    const int mw   = wave >> 2, nw = wave & 3;      // 2 x 4 waves
    const int m0   = blockIdx.y * 64 + mw * 32;     // wave: 32 rows
    const int n0   = blockIdx.x * 256 + nw * 64;    // wave: 64 cols
    const _Float16* Xb = Xt + (size_t)b * NN * CIN;
    _Float16*       Hb = Ht + (size_t)b * NN * H1;

    v8f acc[2][4] = {};
    for (int kk = 0; kk < CIN; kk += 32) {
        v16u a[2], bm[4];
        #pragma unroll
        for (int mi = 0; mi < 2; ++mi) {
            const _Float16* p = W1h + (size_t)(m0 + mi * 16 + lm) * CIN + kk + lh * 8;
            a[mi].h[0] = *(const v8h*)p;
            a[mi].h[1] = *(const v8h*)(p + 16);
        }
        #pragma unroll
        for (int ni = 0; ni < 4; ++ni) {
            const _Float16* p = Xb + (size_t)(n0 + ni * 16 + lm) * CIN + kk + lh * 8;
            bm[ni].h[0] = *(const v8h*)p;
            bm[ni].h[1] = *(const v8h*)(p + 16);
        }
        #pragma unroll
        for (int mi = 0; mi < 2; ++mi)
            #pragma unroll
            for (int ni = 0; ni < 4; ++ni)
                acc[mi][ni] = __builtin_amdgcn_wmma_f32_16x16x32_f16(
                    false, a[mi].v, false, bm[ni].v, (short)0, acc[mi][ni], false, false);
    }
    #pragma unroll
    for (int mi = 0; mi < 2; ++mi) {
        const int mb = m0 + mi * 16 + lh * 8;
        float bv[8];
        #pragma unroll
        for (int v = 0; v < 8; ++v) bv[v] = bias[mb + v];
        #pragma unroll
        for (int ni = 0; ni < 4; ++ni) {
            const int n = n0 + ni * 16 + lm;
            v8h out;
            #pragma unroll
            for (int v = 0; v < 8; ++v) {
                float x = acc[mi][ni][v] + bv[v];
                out[v] = (_Float16)(x > 0.f ? x : 0.f);
            }
            *(v8h*)(Hb + (size_t)n * H1 + mb) = out;
        }
    }
}

// ---------------- kernel 6: GEMM2  out = relu(W2 * H + b2), WMMA f16, f32 out ----------------
__global__ __launch_bounds__(256) void gemm2_kernel(const _Float16* __restrict__ W2h,
                                                    const _Float16* __restrict__ Ht,
                                                    const float*    __restrict__ bias,
                                                    float*          __restrict__ out) {
    const int b    = blockIdx.z;
    const int tid  = threadIdx.x;
    const int wave = tid >> 5, lane = tid & 31;
    const int lm   = lane & 15, lh = lane >> 4;
    const int mw   = wave >> 2, nw = wave & 3;
    const int m0   = blockIdx.y * 64 + mw * 32;     // H2=128 -> blockIdx.y in {0,1}
    const int n0   = blockIdx.x * 256 + nw * 64;
    const _Float16* Hb = Ht + (size_t)b * NN * H1;

    v8f acc[2][4] = {};
    for (int kk = 0; kk < H1; kk += 32) {
        v16u a[2], bm[4];
        #pragma unroll
        for (int mi = 0; mi < 2; ++mi) {
            const _Float16* p = W2h + (size_t)(m0 + mi * 16 + lm) * H1 + kk + lh * 8;
            a[mi].h[0] = *(const v8h*)p;
            a[mi].h[1] = *(const v8h*)(p + 16);
        }
        #pragma unroll
        for (int ni = 0; ni < 4; ++ni) {
            const _Float16* p = Hb + (size_t)(n0 + ni * 16 + lm) * H1 + kk + lh * 8;
            bm[ni].h[0] = *(const v8h*)p;
            bm[ni].h[1] = *(const v8h*)(p + 16);
        }
        #pragma unroll
        for (int mi = 0; mi < 2; ++mi)
            #pragma unroll
            for (int ni = 0; ni < 4; ++ni)
                acc[mi][ni] = __builtin_amdgcn_wmma_f32_16x16x32_f16(
                    false, a[mi].v, false, bm[ni].v, (short)0, acc[mi][ni], false, false);
    }
    #pragma unroll
    for (int mi = 0; mi < 2; ++mi) {
        const int mb = m0 + mi * 16 + lh * 8;
        float bv[8];
        #pragma unroll
        for (int v = 0; v < 8; ++v) bv[v] = bias[mb + v];
        #pragma unroll
        for (int ni = 0; ni < 4; ++ni) {
            const int n = n0 + ni * 16 + lm;
            #pragma unroll
            for (int v = 0; v < 8; ++v) {
                float x = acc[mi][ni][v] + bv[v];
                out[((size_t)b * H2 + mb + v) * NN + n] = x > 0.f ? x : 0.f;
            }
        }
    }
}

// ---------------- launcher ----------------
extern "C" void kernel_launch(void* const* d_in, const int* in_sizes, int n_in,
                              void* d_out, int out_size, void* d_ws, size_t ws_size,
                              hipStream_t stream) {
    (void)in_sizes; (void)n_in; (void)out_size; (void)ws_size;
    const float* unknown = (const float*)d_in[0];  // (B,N,3)
    const float* known   = (const float*)d_in[1];  // (B,M,3)
    const float* uf      = (const float*)d_in[2];  // (B,C1,N)
    const float* kf      = (const float*)d_in[3];  // (B,C2,M)
    const float* W1      = (const float*)d_in[4];  // (H1,CIN)
    const float* b1      = (const float*)d_in[5];  // (H1)
    const float* W2      = (const float*)d_in[6];  // (H2,H1)
    const float* b2      = (const float*)d_in[7];  // (H2)
    float* out = (float*)d_out;                    // (B,H2,N)

    char* ws = (char*)d_ws;
    // workspace layout (16B-aligned offsets)
    int*      idx  = (int*)     (ws + 0);                    //  393216 B
    float*    wgt  = (float*)   (ws + 393216);               //  393216 B
    float*    kft  = (float*)   (ws + 786432);               // 8388608 B  (B,M,C2) f32
    _Float16* W1h  = (_Float16*)(ws + 9175040);              //  196608 B
    _Float16* W2h  = (_Float16*)(ws + 9371648);              //   65536 B
    _Float16* Xt   = (_Float16*)(ws + 9437184);              // 25165824 B (B,N,384) f16
    _Float16* Ht   = (_Float16*)(ws + 34603008);             // 16777216 B (B,N,256) f16
    // total: 51380224 bytes

    convert_w_kernel<<<512, 256, 0, stream>>>(W1, W2, W1h, W2h);
    transpose_kf_kernel<<<(BB * C2 * MM) / 256, 256, 0, stream>>>(kf, kft);
    knn3_kernel<<<dim3(NN / 256, BB), 256, 0, stream>>>(unknown, known, idx, wgt);
    interp_concat_kernel<<<dim3(NN / 128, BB), 128, 0, stream>>>(kft, uf, idx, wgt, Xt);
    gemm1_kernel<<<dim3(NN / 256, H1 / 64, BB), 256, 0, stream>>>(W1h, Xt, b1, Ht);
    gemm2_kernel<<<dim3(NN / 256, H2 / 64, BB), 256, 0, stream>>>(W2h, Ht, b2, out);
}
